// NormalGraphNN_31980326486290
// MI455X (gfx1250) — matrix operand order repaired and checked by reading, not compiled, not run on hardware
//
#include <hip/hip_runtime.h>
#include <hip/hip_bf16.h>

typedef __attribute__((ext_vector_type(16))) __bf16 v16bf;
typedef __attribute__((ext_vector_type(8)))  float  v8f;

// round-to-nearest-even f32 -> bf16
__device__ __forceinline__ __bf16 f2bf(float f) {
    union { float f; unsigned u; } v; v.f = f;
    unsigned r = v.u + 0x7FFFu + ((v.u >> 16) & 1u);
    unsigned short s = (unsigned short)(r >> 16);
    union { unsigned short s; __bf16 b; } o; o.s = s;
    return o.b;
}

__global__ void zero_f32(float* __restrict__ p, long long n) {
    long long i = (long long)blockIdx.x * blockDim.x + threadIdx.x;
    if (i < n) p[i] = 0.0f;
}

__global__ void deg_count(const int* __restrict__ dst, float* __restrict__ deg, int E) {
    int e = blockIdx.x * blockDim.x + threadIdx.x;
    if (e < E) atomicAdd(&deg[dst[e]], 1.0f);
}

__global__ void deg_finalize(float* __restrict__ deg, int Nn) {
    int i = blockIdx.x * blockDim.x + threadIdx.x;
    if (i < Nn) deg[i] = rsqrtf(deg[i] + 1.0f);   // deg buffer becomes deg_inv_sqrt
}

// ---------------------------------------------------------------------------
// Fragment packing: convert fp32 matrices into bf16 laid out exactly in the
// per-lane order the WMMA instruction consumes (ISA 7.12.2, 16-bit A / B).
// Fragment f of a (rowTile,kt) pair occupies 512 contiguous bf16:
//   out[(frag*32 + lane)*16 + p]
// A element (lane,p) -> row = rowTile*16 + (lane&15),
//                       k   = kt*32 + ((p&8)?16:0) + 8*(lane>>4) + (p&7)
// B element (lane,p) -> col = colTile*16 + (lane&15),
//                       k   = kt*32 + ((p&8)?16:0) + 8*(lane>>4) + (p&7)
// ---------------------------------------------------------------------------
template <int K>
__global__ void pack_A(const float* __restrict__ A, __bf16* __restrict__ out, long long total) {
    long long gid = (long long)blockIdx.x * blockDim.x + threadIdx.x;
    if (gid >= total) return;
    constexpr int KT = K / 32;
    int  p    = (int)(gid & 15);
    int  lane = (int)((gid >> 4) & 31);
    long long frag = gid >> 9;
    long long rowTile = frag / KT;
    int  kt   = (int)(frag % KT);
    int  m    = lane & 15, half = lane >> 4;
    int  k    = kt * 32 + ((p & 8) ? 16 : 0) + 8 * half + (p & 7);
    out[gid] = f2bf(A[(rowTile * 16 + m) * K + k]);
}

template <int K, int NO>
__global__ void pack_B(const float* __restrict__ B, __bf16* __restrict__ out) {
    int gid = blockIdx.x * blockDim.x + threadIdx.x;
    if (gid >= K * NO) return;
    constexpr int KT = K / 32;
    int p    = gid & 15;
    int lane = (gid >> 4) & 31;
    int frag = gid >> 9;
    int colTile = frag / KT;
    int kt      = frag % KT;
    int n = lane & 15, half = lane >> 4;
    int k = kt * 32 + ((p & 8) ? 16 : 0) + 8 * half + (p & 7);
    out[gid] = f2bf(B[(size_t)k * NO + colTile * 16 + n]);
}

// C[M,NO] = A @ B from pre-packed bf16 fragments; one wave per 16x16 C tile.
// Inner loop: two 32B coalesced loads + v_wmma_f32_16x16x32_bf16, fully unrolled.
template <int K, int NO>
__global__ void gemm_frag_wmma(const v16bf* __restrict__ Afrag, const v16bf* __restrict__ Bfrag,
                               float* __restrict__ C, int totalTiles) {
    constexpr int KT = K / 32;
    constexpr int colTiles = NO / 16;
    const int tile = blockIdx.x * blockDim.y + threadIdx.y;
    if (tile >= totalTiles) return;          // uniform per wave -> EXEC all-1s for WMMA
    const int rowTile = tile / colTiles;
    const int colTile = tile % colTiles;
    const int lane = threadIdx.x;

    const v16bf* Ap = Afrag + (size_t)rowTile * KT * 32 + lane;
    const v16bf* Bp = Bfrag + (size_t)colTile * KT * 32 + lane;

    v8f acc = {};
#pragma unroll
    for (int kt = 0; kt < KT; ++kt) {
        v16bf a = Ap[kt * 32];
        v16bf b = Bp[kt * 32];
        acc = __builtin_amdgcn_wmma_f32_16x16x32_bf16(
            /*neg_a=*/false, a, /*neg_b=*/false, b,
            /*c_mod=*/(short)0, acc, /*reuse_a=*/false, /*reuse_b=*/false);
    }
    const int half = lane >> 4, n = lane & 15;
#pragma unroll
    for (int j = 0; j < 8; ++j)
        C[(size_t)(rowTile * 16 + 8 * half + j) * NO + colTile * 16 + n] = acc[j];
}

// agg[dst] += h[src] * dis[src]*dis[dst] ; F/4 threads per edge, float4 gathers.
template <int F>
__global__ void scatter_edges(const float* __restrict__ h, const int* __restrict__ src,
                              const int* __restrict__ dst, const float* __restrict__ dis,
                              float* __restrict__ agg, int E) {
    constexpr int FQ = F / 4;
    long long gid = (long long)blockIdx.x * blockDim.x + threadIdx.x;
    int e = (int)(gid / FQ);
    int q = (int)(gid % FQ);
    if (e >= E) return;
    int s = src[e], d = dst[e];
    float nm = dis[s] * dis[d];
    const float4 hv = *(const float4*)(h + (size_t)s * F + 4 * q);
    float* ap = agg + (size_t)d * F + 4 * q;
    atomicAdd(ap + 0, hv.x * nm);
    atomicAdd(ap + 1, hv.y * nm);
    atomicAdd(ap + 2, hv.z * nm);
    atomicAdd(ap + 3, hv.w * nm);
}

// Layer-1 epilogue: x = agg + h*dis^2 + b1, then row L2-normalize. One wave per node (F=256).
__global__ void epi_norm(const float* __restrict__ h, const float* __restrict__ b1,
                         const float* __restrict__ dis, float* __restrict__ agg_x, int Nn) {
    int wid  = threadIdx.x >> 5;
    int lane = threadIdx.x & 31;
    int node = blockIdx.x * (blockDim.x >> 5) + wid;
    if (node >= Nn) return;
    float d2 = dis[node]; d2 *= d2;
    size_t base = (size_t)node * 256;
    float v[8];
    float ss = 0.0f;
#pragma unroll
    for (int j = 0; j < 8; ++j) {
        int f = lane + 32 * j;
        float val = agg_x[base + f] + h[base + f] * d2 + b1[f];
        v[j] = val;
        ss += val * val;
    }
#pragma unroll
    for (int off = 16; off > 0; off >>= 1) ss += __shfl_xor(ss, off, 32);
    float scale = 1.0f / fmaxf(sqrtf(ss), 1e-12f);
#pragma unroll
    for (int j = 0; j < 8; ++j) agg_x[base + lane + 32 * j] = v[j] * scale;
}

// Layer-2 epilogue: out += h2*dis^2 + b2 (out already holds edge aggregation).
__global__ void epi_out(const float* __restrict__ h2, const float* __restrict__ b2,
                        const float* __restrict__ dis, float* __restrict__ out, int total) {
    int idx = blockIdx.x * blockDim.x + threadIdx.x;
    if (idx >= total) return;
    int node = idx >> 7;          // F = 128
    int f    = idx & 127;
    float d = dis[node];
    out[idx] = out[idx] + h2[idx] * d * d + b2[f];
}

extern "C" void kernel_launch(void* const* d_in, const int* in_sizes, int n_in,
                              void* d_out, int out_size, void* d_ws, size_t ws_size,
                              hipStream_t stream) {
    const float* emb = (const float*)d_in[0];
    const float* W1  = (const float*)d_in[1];
    const float* b1  = (const float*)d_in[2];
    const float* W2  = (const float*)d_in[3];
    const float* b2  = (const float*)d_in[4];
    const int*   eix = (const int*)d_in[5];

    const int Nn = in_sizes[0] / 128;   // 100000
    const int E  = in_sizes[5] / 2;     // 1600000
    const int* src = eix;
    const int* dst = eix + E;

    // workspace layout (256B aligned chunks)
    char* ws = (char*)d_ws;
    size_t off = 0;
    auto alloc = [&](size_t bytes) { char* p = ws + off; off += (bytes + 255) & ~(size_t)255; return p; };
    float*  dis  = (float*)alloc((size_t)Nn * 4);            // deg -> deg_inv_sqrt
    float*  h    = (float*)alloc((size_t)Nn * 256 * 4);      // h1, reused as h2
    float*  agg  = (float*)alloc((size_t)Nn * 256 * 4);      // agg1, becomes x
    __bf16* packAX = (__bf16*)alloc((size_t)Nn * 256 * 2);   // packed emb (layer1), packed x (layer2)
    __bf16* packB1 = (__bf16*)alloc((size_t)128 * 256 * 2);
    __bf16* packB2 = (__bf16*)alloc((size_t)256 * 128 * 2);
    float*  out  = (float*)d_out;

    // 1) degrees -> deg_inv_sqrt
    zero_f32<<<(Nn + 255) / 256, 256, 0, stream>>>(dis, Nn);
    deg_count<<<(E + 255) / 256, 256, 0, stream>>>(dst, dis, E);
    deg_finalize<<<(Nn + 255) / 256, 256, 0, stream>>>(dis, Nn);

    // 2) pack emb and W1 into bf16 WMMA fragments, h1 = emb @ W1
    {
        long long tA = (long long)Nn * 128;
        pack_A<128><<<(int)((tA + 255) / 256), 256, 0, stream>>>(emb, packAX, tA);
        pack_B<128, 256><<<(128 * 256 + 255) / 256, 256, 0, stream>>>(W1, packB1);
        int tiles = (Nn / 16) * (256 / 16);
        dim3 blk(32, 8);
        gemm_frag_wmma<128, 256><<<(tiles + 7) / 8, blk, 0, stream>>>(
            (const v16bf*)packAX, (const v16bf*)packB1, h, tiles);
    }

    // 3) agg = sum over edges of norm * h1[src]
    {
        long long nAgg = (long long)Nn * 256;
        zero_f32<<<(int)((nAgg + 255) / 256), 256, 0, stream>>>(agg, nAgg);
        long long threads = (long long)E * (256 / 4);
        scatter_edges<256><<<(int)((threads + 255) / 256), 256, 0, stream>>>(h, src, dst, dis, agg, E);
    }

    // 4) x = normalize(agg + h1*dis^2 + b1)  (in place in agg)
    epi_norm<<<(Nn + 7) / 8, 256, 0, stream>>>(h, b1, dis, agg, Nn);

    // 5) pack x and W2, h2 = x @ W2 (reuse h buffer)
    {
        long long tA = (long long)Nn * 256;
        pack_A<256><<<(int)((tA + 255) / 256), 256, 0, stream>>>(agg, packAX, tA);
        pack_B<256, 128><<<(256 * 128 + 255) / 256, 256, 0, stream>>>(W2, packB2);
        int tiles = (Nn / 16) * (128 / 16);
        dim3 blk(32, 8);
        gemm_frag_wmma<256, 128><<<(tiles + 7) / 8, blk, 0, stream>>>(
            (const v16bf*)packAX, (const v16bf*)packB2, h, tiles);
    }

    // 6) out = edge aggregation of h2, then + h2*dis^2 + b2
    {
        long long nOut = (long long)Nn * 128;
        zero_f32<<<(int)((nOut + 255) / 256), 256, 0, stream>>>(out, nOut);
        long long threads = (long long)E * (128 / 4);
        scatter_edges<128><<<(int)((threads + 255) / 256), 256, 0, stream>>>(h, src, dst, dis, out, E);
        epi_out<<<(int)((nOut + 255) / 256), 256, 0, stream>>>(h, b2, dis, out, (int)nOut);
    }
}
